// Mw_f_ori_77481210020592
// MI455X (gfx1250) — matrix operand order repaired and checked by reading, not compiled, not run on hardware
//
#include <hip/hip_runtime.h>
#include <hip/hip_bf16.h>

typedef __attribute__((ext_vector_type(16))) _Float16 v16h;
typedef __attribute__((ext_vector_type(8)))  float    v8f;

// ---------------------------------------------------------------------------
// WMMA fragment helpers (contiguous k = 16*(lane/16) + s order in A and B, wave32)
// ---------------------------------------------------------------------------

// A fragment: 16x32 f16 tile from row-major A[*, lda] at (row0, k0).
__device__ __forceinline__ v16h load_afrag(const _Float16* A, int lda, int row0, int k0, int lane) {
  int m  = row0 + (lane & 15);
  int kh = (lane >> 4) * 16;
  const _Float16* p = A + (size_t)m * lda + k0;
  v16h a;
#pragma unroll
  for (int v = 0; v < 8; ++v) {
    int kb = 2 * v + kh;
    a[2 * v]     = p[kb];
    a[2 * v + 1] = p[kb + 1];
  }
  return a;
}

// B fragment where B = W^T, W row-major [N, K]: tile (n0, k0) of the 32x16 B.
__device__ __forceinline__ v16h load_bfragT(const _Float16* W, int ldw, int n0, int k0, int lane) {
  int n  = n0 + (lane & 15);
  int kb = (lane >> 4) * 16;
  const _Float16* p = W + (size_t)n * ldw + k0 + kb;
  v16h b;
#pragma unroll
  for (int v = 0; v < 8; ++v) { b[2 * v] = p[2 * v]; b[2 * v + 1] = p[2 * v + 1]; }
  return b;
}

// B fragment from row-major B[K, N] (no transpose): tile (k0, n0).
__device__ __forceinline__ v16h load_bfragKN(const _Float16* Bm, int ldb, int k0, int n0, int lane) {
  int n  = n0 + (lane & 15);
  int kb = k0 + (lane >> 4) * 16;
  v16h b;
#pragma unroll
  for (int v = 0; v < 8; ++v) {
    b[2 * v]     = Bm[(size_t)(kb + 2 * v) * ldb + n];
    b[2 * v + 1] = Bm[(size_t)(kb + 2 * v + 1) * ldb + n];
  }
  return b;
}

__device__ __forceinline__ v8f wmma_f16(v16h a, v16h b, v8f c) {
  return __builtin_amdgcn_wmma_f32_16x16x32_f16(false, a, false, b, (short)0, c, false, false);
}

__device__ __forceinline__ float act_apply(float v, int act) {
  if (act == 1) return tanhf(v);
  if (act == 2) return 1.f / (1.f + __expf(-v));
  if (act == 3) return v > 0.f ? v : 0.01f * v;
  return v;
}

// ---------------------------------------------------------------------------
// Generic batched GEMM: C[b] = A[b] * W[b]^T (+bias, activation)
// A f16 [M, lda], W f16 [Ntiles*16, ldw], C f32 [M, ldc]. M % 16 == 0.
// ---------------------------------------------------------------------------
__global__ void gemm_ta(const _Float16* __restrict__ A, long long sA, int lda,
                        const _Float16* __restrict__ W, long long sW, int ldw,
                        float* __restrict__ C, long long sC, int ldc,
                        int Ntiles, int Nreal, int Ksteps,
                        const float* __restrict__ bias, int act) {
  int wave = threadIdx.x >> 5, lane = threadIdx.x & 31;
  int nt = blockIdx.y * (blockDim.x >> 5) + wave;
  if (nt >= Ntiles) return;                       // wave-uniform
  int mt = blockIdx.x, b = blockIdx.z;
  const _Float16* Ab = A + (size_t)b * sA;
  const _Float16* Wb = W + (size_t)b * sW;
  float*          Cb = C + (size_t)b * sC;
  v8f acc = {0.f, 0.f, 0.f, 0.f, 0.f, 0.f, 0.f, 0.f};
  for (int ks = 0; ks < Ksteps; ++ks) {
    if (ks + 1 < Ksteps)
      __builtin_prefetch(Ab + (size_t)(mt * 16 + (lane & 15)) * lda + (ks + 1) * 32, 0, 1);
    v16h a = load_afrag(Ab, lda, mt * 16, ks * 32, lane);
    v16h w = load_bfragT(Wb, ldw, nt * 16, ks * 32, lane);
    acc = wmma_f16(a, w, acc);
  }
  int n  = nt * 16 + (lane & 15);
  int mb = mt * 16 + (lane >> 4) * 8;
  float bv = (bias && n < Nreal) ? bias[n] : 0.f;
#pragma unroll
  for (int r = 0; r < 8; ++r)
    Cb[(size_t)(mb + r) * ldc + n] = act_apply(acc[r] + bv, act);
}

// C[b] = A[b] * B[b] with row-major B[K, N] (softmax-score application).
__global__ void gemm_ab(const _Float16* __restrict__ A, long long sA, int lda,
                        const _Float16* __restrict__ Bm, long long sB, int ldb,
                        float* __restrict__ C, long long sC, int ldc,
                        int Ntiles, int Ksteps) {
  int wave = threadIdx.x >> 5, lane = threadIdx.x & 31;
  int nt = blockIdx.y * (blockDim.x >> 5) + wave;
  if (nt >= Ntiles) return;
  int mt = blockIdx.x, b = blockIdx.z;
  const _Float16* Ab = A + (size_t)b * sA;
  const _Float16* Bb = Bm + (size_t)b * sB;
  float*          Cb = C + (size_t)b * sC;
  v8f acc = {0.f, 0.f, 0.f, 0.f, 0.f, 0.f, 0.f, 0.f};
  for (int ks = 0; ks < Ksteps; ++ks) {
    v16h a = load_afrag(Ab, lda, mt * 16, ks * 32, lane);
    v16h bb = load_bfragKN(Bb, ldb, ks * 32, nt * 16, lane);
    acc = wmma_f16(a, bb, acc);
  }
  int n  = nt * 16 + (lane & 15);
  int mb = mt * 16 + (lane >> 4) * 8;
#pragma unroll
  for (int r = 0; r < 8; ++r)
    Cb[(size_t)(mb + r) * ldc + n] = acc[r];
}

// ---------------------------------------------------------------------------
// GRU scan (one direction per blockIdx.y; one 16-sequence tile per blockIdx.x)
// h kept in LDS (f16 for WMMA A-operand, f32 for exact recurrence).
// Per step: gh = h * Whh^T via WMMA (Ntiles split over 8 waves), then gates.
// ---------------------------------------------------------------------------
#define GRU_MAXHP 160
#define GRU_MAXN3P 464
__global__ void gru_scan(const float* __restrict__ xw, long long xw_dir_stride, int N3p,
                         const _Float16* __restrict__ Whf, const _Float16* __restrict__ Whb, int Hp,
                         const float* __restrict__ bhf, const float* __restrict__ bhb,
                         float* __restrict__ hs, int ldo, int H, int L) {
  __shared__ _Float16 h16[16 * GRU_MAXHP];
  __shared__ float    h32[16 * GRU_MAXHP];
  __shared__ float    gh[16 * GRU_MAXN3P];
  int dir = blockIdx.y;
  int st  = blockIdx.x;
  const _Float16* Wh = dir ? Whb : Whf;
  const float*    bh = dir ? bhb : bhf;
  const float*   xwp = xw + (size_t)dir * xw_dir_stride;
  int wave = threadIdx.x >> 5, lane = threadIdx.x & 31, nw = blockDim.x >> 5;
  int Ntiles = N3p / 16, Ksteps = Hp / 32;
  for (int i = threadIdx.x; i < 16 * Hp; i += blockDim.x) {
    h16[i] = (_Float16)0.f;
    h32[i] = 0.f;
  }
  __syncthreads();
  for (int t = 0; t < L; ++t) {
    int tt = dir ? (L - 1 - t) : t;
    for (int nt = wave; nt < Ntiles; nt += nw) {
      v8f acc = {0.f, 0.f, 0.f, 0.f, 0.f, 0.f, 0.f, 0.f};
      for (int ks = 0; ks < Ksteps; ++ks) {
        v16h a = load_afrag(h16, Hp, 0, ks * 32, lane);
        v16h w = load_bfragT(Wh, Hp, nt * 16, ks * 32, lane);
        acc = wmma_f16(a, w, acc);
      }
      int n  = nt * 16 + (lane & 15);
      int mb = (lane >> 4) * 8;
#pragma unroll
      for (int r = 0; r < 8; ++r) gh[(mb + r) * N3p + n] = acc[r];
    }
    __syncthreads();
    for (int i = threadIdx.x; i < 16 * H; i += blockDim.x) {
      int s = i / H, j = i % H;
      int seq = st * 16 + s;
      const float* xr = xwp + ((size_t)seq * L + tt) * N3p;
      float gr = gh[s * N3p + j]         + bh[j];
      float gz = gh[s * N3p + H + j]     + bh[H + j];
      float gn = gh[s * N3p + 2 * H + j] + bh[2 * H + j];
      float r = 1.f / (1.f + __expf(-(xr[j] + gr)));
      float z = 1.f / (1.f + __expf(-(xr[H + j] + gz)));
      float n = tanhf(xr[2 * H + j] + r * gn);
      float hn = (1.f - z) * n + z * h32[s * Hp + j];
      h32[s * Hp + j] = hn;
      h16[s * Hp + j] = (_Float16)hn;
      hs[((size_t)seq * L + tt) * ldo + dir * H + j] = hn;
    }
    __syncthreads();
  }
}

// ---------------------------------------------------------------------------
// Fused elementwise-mult / elementwise-sub attention scores.
// blockIdx.x = b*Lq + q; blockIdx.y = 0 (mult -> Sd) / 1 (sub -> Sm).
// X[p,k] = op(hp[b,q,k], hq[b,p,k]); S[p] = v . tanh(X * W^T)
// ---------------------------------------------------------------------------
__global__ void multsub_att(const _Float16* __restrict__ hq_h /*[B,416,256]*/,
                            const _Float16* __restrict__ hp_h /*[B*Lq,256]*/,
                            const _Float16* __restrict__ Wd_h, const _Float16* __restrict__ Wm_h,
                            const float* __restrict__ vd, const float* __restrict__ vm,
                            float* __restrict__ Sd, float* __restrict__ Sm,
                            int Lp, int Lq) {
  int row = blockIdx.x, mode = blockIdx.y;
  int b = row / Lq;
  const _Float16* Wsel = mode ? Wm_h : Wd_h;
  const float*    vsel = mode ? vm : vd;
  float*          Ssel = mode ? Sm : Sd;
  __shared__ _Float16 hpv[256];
  for (int i = threadIdx.x; i < 256; i += blockDim.x) hpv[i] = hp_h[(size_t)row * 256 + i];
  __syncthreads();
  int wave = threadIdx.x >> 5, lane = threadIdx.x & 31, nw = blockDim.x >> 5;
  const _Float16* hqb = hq_h + (size_t)b * 416 * 256;
  for (int pt = wave; pt < Lp / 16; pt += nw) {
    float sacc[8];
#pragma unroll
    for (int r = 0; r < 8; ++r) sacc[r] = 0.f;
    for (int ntile = 0; ntile < 8; ++ntile) {
      v8f acc = {0.f, 0.f, 0.f, 0.f, 0.f, 0.f, 0.f, 0.f};
      for (int ks = 0; ks < 8; ++ks) {
        int m  = pt * 16 + (lane & 15);
        int kh = (lane >> 4) * 16;
        const _Float16* hqr = hqb + (size_t)m * 256 + ks * 32;
        v16h a;
#pragma unroll
        for (int v = 0; v < 8; ++v) {
          int kb = 2 * v + kh;
          _Float16 x0 = hqr[kb], x1 = hqr[kb + 1];
          _Float16 h0 = hpv[ks * 32 + kb], h1 = hpv[ks * 32 + kb + 1];
          a[2 * v]     = mode ? (_Float16)(h0 - x0) : (_Float16)(h0 * x0);
          a[2 * v + 1] = mode ? (_Float16)(h1 - x1) : (_Float16)(h1 * x1);
        }
        v16h w = load_bfragT(Wsel, 256, ntile * 16, ks * 32, lane);
        acc = wmma_f16(a, w, acc);
      }
      float vn = vsel[ntile * 16 + (lane & 15)];
#pragma unroll
      for (int r = 0; r < 8; ++r) {
        float t = vn * tanhf(acc[r]);
        t += __shfl_xor(t, 1);
        t += __shfl_xor(t, 2);
        t += __shfl_xor(t, 4);
        t += __shfl_xor(t, 8);
        sacc[r] += t;                      // full sum within each 16-lane half
      }
    }
    if ((lane & 15) == 0) {
      int mb = pt * 16 + (lane >> 4) * 8;
#pragma unroll
      for (int r = 0; r < 8; ++r) {
        int p = mb + r;
        if (p < Lp) Ssel[(size_t)row * Lp + p] = sacc[r];
      }
    }
  }
}

// ---------------------------------------------------------------------------
// Glue kernels
// ---------------------------------------------------------------------------
__global__ void pack3(_Float16* __restrict__ dst, const float* __restrict__ src,
                      int Bn, int Rp, int Cp, int R, int C) {
  long long idx = (long long)blockIdx.x * blockDim.x + threadIdx.x;
  long long tot = (long long)Bn * Rp * Cp;
  if (idx >= tot) return;
  int c = (int)(idx % Cp);
  long long t = idx / Cp;
  int r = (int)(t % Rp);
  int b = (int)(t / Rp);
  float v = (r < R && c < C) ? src[((size_t)b * R + r) * C + c] : 0.f;
  dst[idx] = (_Float16)v;
}

__global__ void embed_gather(const int* __restrict__ tok, const float* __restrict__ emb,
                             _Float16* __restrict__ out, int Ntok, int Cp, int D) {
  long long idx = (long long)blockIdx.x * blockDim.x + threadIdx.x;
  if (idx >= (long long)Ntok * Cp) return;
  int c = (int)(idx % Cp), n = (int)(idx / Cp);
  float v = (c < D) ? emb[(size_t)tok[n] * D + c] : 0.f;
  out[idx] = (_Float16)v;
}

// in-place row softmax; optional f16 zero-padded copy.
__global__ void softmax_rows(float* __restrict__ S, _Float16* __restrict__ Sh,
                             int nrows, int L, int Lpad) {
  int wave = threadIdx.x >> 5, lane = threadIdx.x & 31;
  int row = blockIdx.x * (blockDim.x >> 5) + wave;
  if (row >= nrows) return;
  float* r = S + (size_t)row * L;
  float mx = -1e30f;
  for (int i = lane; i < L; i += 32) mx = fmaxf(mx, r[i]);
  for (int m = 16; m; m >>= 1) mx = fmaxf(mx, __shfl_xor(mx, m));
  float sum = 0.f;
  for (int i = lane; i < L; i += 32) { float e = __expf(r[i] - mx); r[i] = e; sum += e; }
  for (int m = 16; m; m >>= 1) sum += __shfl_xor(sum, m);
  float inv = 1.f / sum;
  for (int i = lane; i < L; i += 32) {
    float v = r[i] * inv;
    r[i] = v;
    if (Sh) Sh[(size_t)row * Lpad + i] = (_Float16)v;
  }
  if (Sh)
    for (int i = L + lane; i < Lpad; i += 32) Sh[(size_t)row * Lpad + i] = (_Float16)0.f;
}

__global__ void concat_score(const float* __restrict__ s1, const float* __restrict__ s2,
                             const float* __restrict__ vc, float* __restrict__ Sc,
                             int BLq, int Lq, int Lp, int E) {
  long long idx = (long long)blockIdx.x * blockDim.x + threadIdx.x;
  if (idx >= (long long)BLq * Lp) return;
  int p = (int)(idx % Lp);
  int row = (int)(idx / Lp);
  int b = row / Lq;
  const float* r1 = s1 + ((size_t)b * Lp + p) * E;
  const float* r2 = s2 + (size_t)row * E;
  float s = 0.f;
  for (int e = 0; e < E; ++e) s += vc[e] * tanhf(r1[e] + r2[e]);
  Sc[idx] = s;
}

__global__ void concat_qt_hp(const float* __restrict__ qt, const float* __restrict__ hp,
                             _Float16* __restrict__ xh, int rows) {
  long long idx = (long long)blockIdx.x * blockDim.x + threadIdx.x;
  if (idx >= (long long)rows * 512) return;
  int c = (int)(idx % 512), r = (int)(idx / 512);
  float v = (c < 256) ? qt[(size_t)r * 256 + c] : hp[(size_t)r * 256 + (c - 256)];
  xh[idx] = (_Float16)v;
}

__global__ void gate_mul(const float* __restrict__ g, const float* __restrict__ qt,
                         const float* __restrict__ hp, _Float16* __restrict__ gxh, int rows) {
  long long idx = (long long)blockIdx.x * blockDim.x + threadIdx.x;
  if (idx >= (long long)rows * 512) return;
  int c = (int)(idx % 512), r = (int)(idx / 512);
  float x = (c < 256) ? qt[(size_t)r * 256 + c] : hp[(size_t)r * 256 + (c - 256)];
  gxh[idx] = (_Float16)(g[idx] * x);
}

__global__ void vdot(const float* __restrict__ X, const float* __restrict__ v,
                     float* __restrict__ out, int rows, int E, int ld, int act) {
  int i = blockIdx.x * blockDim.x + threadIdx.x;
  if (i >= rows) return;
  const float* x = X + (size_t)i * ld;
  float s = 0.f;
  for (int e = 0; e < E; ++e) s += v[e] * (act ? tanhf(x[e]) : x[e]);
  out[i] = s;
}

__global__ void agg_softmax(const float* __restrict__ sa /*[4][rows]*/,
                            float* __restrict__ w /*[rows][4]*/, int rows) {
  int i = blockIdx.x * blockDim.x + threadIdx.x;
  if (i >= rows) return;
  float l[4], m = -1e30f, sum = 0.f;
  for (int k = 0; k < 4; ++k) { l[k] = sa[(size_t)k * rows + i]; m = fmaxf(m, l[k]); }
  for (int k = 0; k < 4; ++k) { l[k] = __expf(l[k] - m); sum += l[k]; }
  for (int k = 0; k < 4; ++k) w[(size_t)i * 4 + k] = l[k] / sum;
}

__global__ void agg_rep(const float* __restrict__ w, const float* __restrict__ h0,
                        const float* __restrict__ h1, const float* __restrict__ h2,
                        const float* __restrict__ h3, float* __restrict__ rep, int rows) {
  long long idx = (long long)blockIdx.x * blockDim.x + threadIdx.x;
  if (idx >= (long long)rows * 256) return;
  int i = (int)(idx / 256);
  const float* ww = w + (size_t)i * 4;
  rep[idx] = ww[0] * h0[idx] + ww[1] * h1[idx] + ww[2] * h2[idx] + ww[3] * h3[idx];
}

__global__ void wsum(const float* __restrict__ w, const float* __restrict__ X,
                     float* __restrict__ out, int R, int Lr, int E) {
  long long idx = (long long)blockIdx.x * blockDim.x + threadIdx.x;
  if (idx >= (long long)R * E) return;
  int r = (int)(idx / E), e = (int)(idx % E);
  float s = 0.f;
  for (int l = 0; l < Lr; ++l) s += w[(size_t)r * Lr + l] * X[((size_t)r * Lr + l) * E + e];
  out[idx] = s;
}

__global__ void sp_pre_k(const float* __restrict__ t1, const float* __restrict__ t2,
                         const float* __restrict__ vp, float* __restrict__ out, int rows, int Lq) {
  int i = blockIdx.x * blockDim.x + threadIdx.x;
  if (i >= rows) return;
  int b = i / Lq;
  float s = 0.f;
  for (int e = 0; e < 128; ++e) s += vp[e] * (t1[(size_t)i * 128 + e] + t2[(size_t)b * 128 + e]);
  out[i] = s;
}

__global__ void final_score(const float* __restrict__ a_rep /*[48,300]*/,
                            const float* __restrict__ enc /*[16,304]*/,
                            float* __restrict__ out /*[48]*/) {
  __shared__ float logits[48];
  int idx = threadIdx.x;
  if (idx < 48) {
    int b = idx / 3;
    float s = 0.f;
    for (int d = 0; d < 300; ++d) s += a_rep[(size_t)idx * 300 + d] * enc[(size_t)b * 304 + d];
    logits[idx] = s;
  }
  __syncthreads();
  if (idx < 48) {
    int b = idx / 3;
    float l0 = logits[b * 3], l1 = logits[b * 3 + 1], l2 = logits[b * 3 + 2];
    float m = fmaxf(l0, fmaxf(l1, l2));
    float e0 = __expf(l0 - m), e1 = __expf(l1 - m), e2 = __expf(l2 - m);
    out[idx] = __expf(logits[idx] - m) / (e0 + e1 + e2);
  }
}

// ---------------------------------------------------------------------------
// Host orchestration
// ---------------------------------------------------------------------------
static inline void launch_gemm_ta(hipStream_t s, const _Float16* A, long long sA, int lda,
                                  const _Float16* W, long long sW, int ldw,
                                  float* C, long long sC, int ldc,
                                  int M, int Ntiles, int Nreal, int Ksteps,
                                  const float* bias, int act, int batch) {
  dim3 grid(M / 16, (Ntiles + 7) / 8, batch);
  gemm_ta<<<grid, 256, 0, s>>>(A, sA, lda, W, sW, ldw, C, sC, ldc, Ntiles, Nreal, Ksteps, bias, act);
}

static inline void launch_pack3(hipStream_t s, _Float16* dst, const float* src,
                                int Bn, int Rp, int Cp, int R, int C) {
  long long tot = (long long)Bn * Rp * Cp;
  pack3<<<(unsigned)((tot + 255) / 256), 256, 0, s>>>(dst, src, Bn, Rp, Cp, R, C);
}

// flat input indices (setup_inputs dict order)
enum {
  IN_QUERY = 0, IN_PASSAGE, IN_ANSWER, IN_EMB,
  IN_AGRU = 4, IN_QGRU = 12, IN_PGRU = 20, IN_AGGGRU = 28,
  IN_HTC = 36, IN_HTB = 44, IN_HTD = 52, IN_HTM = 60,
  IN_AATT = 68, IN_WC1, IN_WC2, IN_VC, IN_WB, IN_WD, IN_VD, IN_WM, IN_VM,
  IN_WGC, IN_WGB, IN_WGD, IN_WGM,
  IN_WAGG, IN_VAGG, IN_WQ, IN_VQ, IN_WP1, IN_WP2, IN_VP, IN_PRED
};

struct GruW {
  const _Float16 *Wif, *Whf, *Wib, *Whb;
  const float *bif, *bhf, *bib, *bhb;
  int H, Hp, N3, N3p, Inp;
};

extern "C" void kernel_launch(void* const* d_in, const int* in_sizes, int n_in,
                              void* d_out, int out_size, void* d_ws, size_t ws_size,
                              hipStream_t stream) {
  const int B = 16, Lq = 64, Lp = 400, La = 16, D = 300, Dp = 320;
  const int E = 128, H2 = 256, Lpp = 416;
  (void)in_sizes; (void)n_in; (void)out_size; (void)ws_size;

  size_t off = 0;
  auto alloc = [&](size_t bytes) -> void* {
    void* p = (char*)d_ws + off;
    off = (off + bytes + 255) & ~(size_t)255;
    return p;
  };
  auto allocF  = [&](size_t n) { return (float*)alloc(n * sizeof(float)); };
  auto allocH  = [&](size_t n) { return (_Float16*)alloc(n * sizeof(_Float16)); };

  auto pack_w = [&](int idx, int N, int K, int Np, int Kp) -> _Float16* {
    _Float16* d = allocH((size_t)Np * Kp);
    launch_pack3(stream, d, (const float*)d_in[idx], 1, Np, Kp, N, K);
    return d;
  };
  auto pack_gru = [&](int base, int In, int H) -> GruW {
    GruW g;
    g.H = H; g.N3 = 3 * H;
    g.N3p = (g.N3 + 15) / 16 * 16;
    g.Hp  = (H + 31) / 32 * 32;
    g.Inp = (In + 31) / 32 * 32;
    g.Wif = pack_w(base + 0, g.N3, In, g.N3p, g.Inp);
    g.Whf = pack_w(base + 1, g.N3, H,  g.N3p, g.Hp);
    g.bif = (const float*)d_in[base + 2];
    g.bhf = (const float*)d_in[base + 3];
    g.Wib = pack_w(base + 4, g.N3, In, g.N3p, g.Inp);
    g.Whb = pack_w(base + 5, g.N3, H,  g.N3p, g.Hp);
    g.bib = (const float*)d_in[base + 6];
    g.bhb = (const float*)d_in[base + 7];
    return g;
  };

  // shared scratch for GRU input projections (max: q_gru 2*6400*384 floats)
  float* xw_arena = allocF((size_t)2 * 6400 * 464);
  auto run_bigru = [&](const _Float16* x_h, int M, const GruW& g, int Nseq, int L, float* hs) {
    long long ds = (long long)M * g.N3p;
    launch_gemm_ta(stream, x_h, 0, g.Inp, g.Wif, 0, g.Inp, xw_arena, 0, g.N3p,
                   M, g.N3p / 16, g.N3, g.Inp / 32, g.bif, 0, 1);
    launch_gemm_ta(stream, x_h, 0, g.Inp, g.Wib, 0, g.Inp, xw_arena + ds, 0, g.N3p,
                   M, g.N3p / 16, g.N3, g.Inp / 32, g.bib, 0, 1);
    gru_scan<<<dim3(Nseq / 16, 2), 256, 0, stream>>>(xw_arena, ds, g.N3p, g.Whf, g.Whb, g.Hp,
                                                     g.bhf, g.bhb, hs, 2 * g.H, g.H, L);
  };

  const float* emb = (const float*)d_in[IN_EMB];

  // ---- embeddings -> f16 padded ----
  _Float16* qe_h = allocH((size_t)B * Lq * Dp);
  _Float16* pe_h = allocH((size_t)B * Lp * Dp);
  _Float16* ae_h = allocH((size_t)B * 3 * La * Dp);
  embed_gather<<<(B * Lq * Dp + 255) / 256, 256, 0, stream>>>((const int*)d_in[IN_QUERY], emb, qe_h, B * Lq, Dp, D);
  embed_gather<<<(B * Lp * Dp + 255) / 256, 256, 0, stream>>>((const int*)d_in[IN_PASSAGE], emb, pe_h, B * Lp, Dp, D);
  embed_gather<<<(B * 3 * La * Dp + 255) / 256, 256, 0, stream>>>((const int*)d_in[IN_ANSWER], emb, ae_h, B * 3 * La, Dp, D);

  // ---- answer path: a_gru (H=150), attention pooling ----
  GruW ga = pack_gru(IN_AGRU, D, 150);
  float* a_h = allocF((size_t)48 * La * 300);
  run_bigru(ae_h, 48 * La, ga, 48, La, a_h);
  float* aw = allocF(48 * La);
  vdot<<<(48 * La + 255) / 256, 256, 0, stream>>>(a_h, (const float*)d_in[IN_AATT], aw, 48 * La, 300, 300, 0);
  softmax_rows<<<(48 + 7) / 8, 256, 0, stream>>>(aw, nullptr, 48, La, 0);
  float* a_rep = allocF(48 * 300);
  wsum<<<(48 * 300 + 255) / 256, 256, 0, stream>>>(aw, a_h, a_rep, 48, La, 300);

  // ---- main encoders (note reference swaps them) ----
  GruW gq = pack_gru(IN_QGRU, D, E);
  GruW gp = pack_gru(IN_PGRU, D, E);
  float* hq = allocF((size_t)B * Lp * H2);   // bigru(pe, q_gru)
  float* hp = allocF((size_t)B * Lq * H2);   // bigru(qe, p_gru)
  run_bigru(pe_h, B * Lp, gq, B, Lp, hq);
  run_bigru(qe_h, B * Lq, gp, B, Lq, hp);

  _Float16* hq_h = allocH((size_t)B * Lpp * H2);   // row-padded per batch for K=416 GEMMs
  _Float16* hp_h = allocH((size_t)B * Lq * H2);
  launch_pack3(stream, hq_h, hq, B, Lpp, H2, Lp, H2);
  launch_pack3(stream, hp_h, hp, 1, B * Lq, H2, B * Lq, H2);

  // ---- concat (additive) attention ----
  _Float16* Wc1_h = pack_w(IN_WC1, E, H2, E, H2);
  _Float16* Wc2_h = pack_w(IN_WC2, E, H2, E, H2);
  float* s1 = allocF((size_t)B * Lp * E);
  float* s2 = allocF((size_t)B * Lq * E);
  launch_gemm_ta(stream, hq_h, (long long)Lpp * H2, H2, Wc1_h, 0, H2,
                 s1, (long long)Lp * E, E, Lp, E / 16, E, H2 / 32, nullptr, 0, B);
  launch_gemm_ta(stream, hp_h, 0, H2, Wc2_h, 0, H2,
                 s2, 0, E, B * Lq, E / 16, E, H2 / 32, nullptr, 0, 1);
  float* Sc = allocF((size_t)B * Lq * Lp);
  concat_score<<<(B * Lq * Lp + 255) / 256, 256, 0, stream>>>(s1, s2, (const float*)d_in[IN_VC], Sc, B * Lq, Lq, Lp, E);
  _Float16* Sc_h = allocH((size_t)B * Lq * Lpp);
  softmax_rows<<<(B * Lq + 7) / 8, 256, 0, stream>>>(Sc, Sc_h, B * Lq, Lp, Lpp);
  float* qtc = allocF((size_t)B * Lq * H2);
  gemm_ab<<<dim3(Lq / 16, 2, B), 256, 0, stream>>>(Sc_h, (long long)Lq * Lpp, Lpp,
      hq_h, (long long)Lpp * H2, H2, qtc, (long long)Lq * H2, H2, H2 / 16, Lpp / 32);

  // ---- bilinear attention ----
  _Float16* Wb_h = pack_w(IN_WB, H2, H2, H2, H2);
  float* tmpb = allocF((size_t)B * Lp * H2);
  launch_gemm_ta(stream, hq_h, (long long)Lpp * H2, H2, Wb_h, 0, H2,
                 tmpb, (long long)Lp * H2, H2, Lp, H2 / 16, H2, H2 / 32, nullptr, 0, B);
  _Float16* tmpb_h = allocH((size_t)B * Lp * H2);
  launch_pack3(stream, tmpb_h, tmpb, 1, B * Lp, H2, B * Lp, H2);
  float* Sb = allocF((size_t)B * Lq * Lp);
  launch_gemm_ta(stream, hp_h, (long long)Lq * H2, H2, tmpb_h, (long long)Lp * H2, H2,
                 Sb, (long long)Lq * Lp, Lp, Lq, Lp / 16, Lp, H2 / 32, nullptr, 0, B);
  _Float16* Sb_h = allocH((size_t)B * Lq * Lpp);
  softmax_rows<<<(B * Lq + 7) / 8, 256, 0, stream>>>(Sb, Sb_h, B * Lq, Lp, Lpp);
  float* qtb = allocF((size_t)B * Lq * H2);
  gemm_ab<<<dim3(Lq / 16, 2, B), 256, 0, stream>>>(Sb_h, (long long)Lq * Lpp, Lpp,
      hq_h, (long long)Lpp * H2, H2, qtb, (long long)Lq * H2, H2, H2 / 16, Lpp / 32);

  // ---- elementwise mult / sub attentions (fused WMMA) ----
  _Float16* Wd_h = pack_w(IN_WD, E, H2, E, H2);
  _Float16* Wm_h = pack_w(IN_WM, E, H2, E, H2);
  float* Sd = allocF((size_t)B * Lq * Lp);
  float* Sm = allocF((size_t)B * Lq * Lp);
  multsub_att<<<dim3(B * Lq, 2), 256, 0, stream>>>(hq_h, hp_h, Wd_h, Wm_h,
      (const float*)d_in[IN_VD], (const float*)d_in[IN_VM], Sd, Sm, Lp, Lq);
  _Float16* Sd_h = allocH((size_t)B * Lq * Lpp);
  _Float16* Sm_h = allocH((size_t)B * Lq * Lpp);
  softmax_rows<<<(B * Lq + 7) / 8, 256, 0, stream>>>(Sd, Sd_h, B * Lq, Lp, Lpp);
  softmax_rows<<<(B * Lq + 7) / 8, 256, 0, stream>>>(Sm, Sm_h, B * Lq, Lp, Lpp);
  float* qtd = allocF((size_t)B * Lq * H2);
  float* qtm = allocF((size_t)B * Lq * H2);
  gemm_ab<<<dim3(Lq / 16, 2, B), 256, 0, stream>>>(Sd_h, (long long)Lq * Lpp, Lpp,
      hq_h, (long long)Lpp * H2, H2, qtd, (long long)Lq * H2, H2, H2 / 16, Lpp / 32);
  gemm_ab<<<dim3(Lq / 16, 2, B), 256, 0, stream>>>(Sm_h, (long long)Lq * Lpp, Lpp,
      hq_h, (long long)Lpp * H2, H2, qtm, (long long)Lq * H2, H2, H2 / 16, Lpp / 32);

  // ---- gated aggregation GRUs (4 views) ----
  const int rows = B * Lq;
  _Float16* xcat_h = allocH((size_t)rows * 512);
  float*    gbuf   = allocF((size_t)rows * 512);
  _Float16* gx_h   = allocH((size_t)rows * 512);
  float*     qts[4] = {qtc, qtb, qtd, qtm};
  int        wgs[4] = {IN_WGC, IN_WGB, IN_WGD, IN_WGM};
  int        grub[4] = {IN_HTC, IN_HTB, IN_HTD, IN_HTM};
  float*     ht[4];
  _Float16*  ht_h[4];
  for (int k = 0; k < 4; ++k) {
    _Float16* Wg_h = pack_w(wgs[k], 512, 512, 512, 512);
    concat_qt_hp<<<(rows * 512 + 255) / 256, 256, 0, stream>>>(qts[k], hp, xcat_h, rows);
    launch_gemm_ta(stream, xcat_h, 0, 512, Wg_h, 0, 512, gbuf, 0, 512,
                   rows, 512 / 16, 512, 512 / 32, nullptr, 2, 1);
    gate_mul<<<(rows * 512 + 255) / 256, 256, 0, stream>>>(gbuf, qts[k], hp, gx_h, rows);
    GruW gg = pack_gru(grub[k], 512, E);
    ht[k] = allocF((size_t)rows * H2);
    run_bigru(gx_h, rows, gg, B, Lq, ht[k]);
    ht_h[k] = allocH((size_t)rows * H2);
    launch_pack3(stream, ht_h[k], ht[k], 1, rows, H2, rows, H2);
  }

  // ---- per-position attention over 4 views ----
  _Float16* Wagg_h = pack_w(IN_WAGG, E, H2, E, H2);
  float* tk = allocF((size_t)4 * rows * E);
  for (int k = 0; k < 4; ++k)
    launch_gemm_ta(stream, ht_h[k], 0, H2, Wagg_h, 0, H2, tk + (size_t)k * rows * E, 0, E,
                   rows, E / 16, E, H2 / 32, nullptr, 0, 1);
  float* sa = allocF((size_t)4 * rows);
  vdot<<<(4 * rows + 255) / 256, 256, 0, stream>>>(tk, (const float*)d_in[IN_VAGG], sa, 4 * rows, E, E, 1);
  float* saw = allocF((size_t)rows * 4);
  agg_softmax<<<(rows + 255) / 256, 256, 0, stream>>>(sa, saw, rows);
  float* rep = allocF((size_t)rows * H2);
  agg_rep<<<(rows * H2 + 255) / 256, 256, 0, stream>>>(saw, ht[0], ht[1], ht[2], ht[3], rep, rows);
  _Float16* rep_h = allocH((size_t)rows * H2);
  launch_pack3(stream, rep_h, rep, 1, rows, H2, rows, H2);
  GruW gagg = pack_gru(IN_AGGGRU, H2, E);
  float* aggr = allocF((size_t)rows * H2);
  run_bigru(rep_h, rows, gagg, B, Lq, aggr);
  _Float16* aggr_h = allocH((size_t)rows * H2);
  launch_pack3(stream, aggr_h, aggr, 1, rows, H2, rows, H2);

  // ---- prediction layer ----
  _Float16* Wq_h = pack_w(IN_WQ, E, H2, E, H2);
  float* tq = allocF((size_t)B * Lp * E);
  launch_gemm_ta(stream, hq_h, (long long)Lpp * H2, H2, Wq_h, 0, H2,
                 tq, (long long)Lp * E, E, Lp, E / 16, E, H2 / 32, nullptr, 0, B);
  float* sqv = allocF((size_t)B * Lp);
  vdot<<<(B * Lp + 255) / 256, 256, 0, stream>>>(tq, (const float*)d_in[IN_VQ], sqv, B * Lp, E, E, 1);
  softmax_rows<<<(B + 7) / 8, 256, 0, stream>>>(sqv, nullptr, B, Lp, 0);
  float* rq = allocF((size_t)B * H2);
  wsum<<<(B * H2 + 255) / 256, 256, 0, stream>>>(sqv, hq, rq, B, Lp, H2);
  _Float16* rq_h = allocH((size_t)B * H2);
  launch_pack3(stream, rq_h, rq, 1, B, H2, B, H2);

  _Float16* Wp1_h = pack_w(IN_WP1, E, H2, E, H2);
  _Float16* Wp2_h = pack_w(IN_WP2, E, H2, E, H2);
  float* t1 = allocF((size_t)rows * E);
  float* t2 = allocF((size_t)B * E);
  launch_gemm_ta(stream, aggr_h, 0, H2, Wp1_h, 0, H2, t1, 0, E, rows, E / 16, E, H2 / 32, nullptr, 0, 1);
  launch_gemm_ta(stream, rq_h, 0, H2, Wp2_h, 0, H2, t2, 0, E, B, E / 16, E, H2 / 32, nullptr, 0, 1);
  float* spv = allocF((size_t)rows);
  sp_pre_k<<<(rows + 255) / 256, 256, 0, stream>>>(t1, t2, (const float*)d_in[IN_VP], spv, rows, Lq);
  softmax_rows<<<(B + 7) / 8, 256, 0, stream>>>(spv, nullptr, B, Lq, 0);
  float* rp = allocF((size_t)B * H2);
  wsum<<<(B * H2 + 255) / 256, 256, 0, stream>>>(spv, aggr, rp, B, Lq, H2);
  _Float16* rp_h = allocH((size_t)B * H2);
  launch_pack3(stream, rp_h, rp, 1, B, H2, B, H2);

  _Float16* pred_h = pack_w(IN_PRED, D, H2, 304, H2);
  float* enc = allocF((size_t)B * 304);
  launch_gemm_ta(stream, rp_h, 0, H2, pred_h, 0, H2, enc, 0, 304,
                 B, 304 / 16, D, H2 / 32, nullptr, 3, 1);     // leaky_relu

  final_score<<<1, 64, 0, stream>>>(a_rep, enc, (float*)d_out);
}
